// FlashAttentionRope_23295902614143
// MI455X (gfx1250) — matrix operand order repaired and checked by reading, not compiled, not run on hardware
//
#include <hip/hip_runtime.h>
#include <hip/hip_bf16.h>

// Problem constants (match reference)
#define B_    2
#define N_    1024
#define C_    1024
#define H_    16
#define CH_   64
#define SP_   3072
#define KSEL_ 4096

typedef _Float16 v16h __attribute__((ext_vector_type(16)));
typedef float    v8f  __attribute__((ext_vector_type(8)));

// ---------------------------------------------------------------------------
// WMMA helpers (CDNA5 gfx1250: V_WMMA_F32_16X16X32_F16, wave32)
// ---------------------------------------------------------------------------
__device__ __forceinline__ v8f wmma_f16(v16h a, v16h b, v8f c) {
  // (neg_a, A, neg_b, B, c_mod, C, reuse_a, reuse_b)
  return __builtin_amdgcn_wmma_f32_16x16x32_f16(false, a, false, b, (short)0, c,
                                                false, false);
}

// 16-bit A-matrix 16x32 layout (ISA 7.12.2):
//   lanes 0-15 : M=lane,   VGPR j holds K = {0..7, 16..23} pairs
//   lanes 16-31: M=lane-16, same + 8
__device__ __forceinline__ int kmapA(int j, int hi) {
  return ((j < 4) ? (2 * j) : (8 + 2 * j)) + (hi << 3);
}
// 16-bit B-matrix 32x16 layout (same K order as A):
//   lanes 0-15 : N=lane,   VGPR j holds K = {0..7, 16..23} pairs
//   lanes 16-31: N=lane-16, same + 8
__device__ __forceinline__ int kmapB(int j, int hi) {
  return ((j < 4) ? (2 * j) : (8 + 2 * j)) + (hi << 3);
}

__device__ __forceinline__ v16h load_a_frag(const _Float16* __restrict__ base,
                                            int ld, int row, int kb, int hi) {
  v16h f;
  const _Float16* p = base + (size_t)row * ld + kb;
#pragma unroll
  for (int j = 0; j < 8; ++j) {
    int k = kmapA(j, hi);
    f[2 * j]     = p[k];
    f[2 * j + 1] = p[k + 1];
  }
  return f;
}

__device__ __forceinline__ v16h load_b_frag(const _Float16* __restrict__ base,
                                            int ld, int row, int kb, int hi) {
  v16h f;
  const _Float16* p = base + (size_t)row * ld + kb;
#pragma unroll
  for (int j = 0; j < 8; ++j) {
    int k = kmapB(j, hi);
    f[2 * j]     = p[k];
    f[2 * j + 1] = p[k + 1];
  }
  return f;
}

// ---------------------------------------------------------------------------
// f32 -> f16 convert (grid-stride)
// ---------------------------------------------------------------------------
__global__ __launch_bounds__(256) void f32_to_f16_kernel(
    const float* __restrict__ src, _Float16* __restrict__ dst, int n) {
  int i = blockIdx.x * blockDim.x + threadIdx.x;
  int stride = gridDim.x * blockDim.x;
  for (; i < n; i += stride) dst[i] = (_Float16)src[i];
}

// ---------------------------------------------------------------------------
// Generic GEMM: C(MxNout) = A(MxK) @ Bw(NoutxK)^T  (+bias), f16 in, f32 out.
// One wave computes a 16x64 strip: 1 A-fragment reused by 4 B-fragments/step.
// Software pipelined: next step's fragments are loaded before current WMMAs.
// ---------------------------------------------------------------------------
__global__ __launch_bounds__(256) void gemm_wmma_kernel(
    const _Float16* __restrict__ A, const _Float16* __restrict__ Bw,
    const float* __restrict__ bias, float* __restrict__ Cout,
    int M, int Nout, int K) {
  const int lane = threadIdx.x & 31;
  const int wave = threadIdx.x >> 5;
  const int hi = lane >> 4;
  const int ln = lane & 15;

  const int tiles_m = M >> 4;
  const int gid = blockIdx.x * 8 + wave;
  const int tm = gid % tiles_m;
  const int tn = gid / tiles_m;  // 64-wide column strip
  if (tn * 64 >= Nout) return;   // uniform per wave

  const int rowBase = tm * 16;
  const int colBase = tn * 64;

  v8f acc0 = {}, acc1 = {}, acc2 = {}, acc3 = {};

  v16h a  = load_a_frag(A, K, rowBase + ln, 0, hi);
  v16h b0 = load_b_frag(Bw, K, colBase + ln, 0, hi);
  v16h b1 = load_b_frag(Bw, K, colBase + 16 + ln, 0, hi);
  v16h b2 = load_b_frag(Bw, K, colBase + 32 + ln, 0, hi);
  v16h b3 = load_b_frag(Bw, K, colBase + 48 + ln, 0, hi);

  for (int kb = 32; kb <= K - 32; kb += 32) {
    // prefetch next K-step
    v16h an  = load_a_frag(A, K, rowBase + ln, kb, hi);
    v16h bn0 = load_b_frag(Bw, K, colBase + ln, kb, hi);
    v16h bn1 = load_b_frag(Bw, K, colBase + 16 + ln, kb, hi);
    v16h bn2 = load_b_frag(Bw, K, colBase + 32 + ln, kb, hi);
    v16h bn3 = load_b_frag(Bw, K, colBase + 48 + ln, kb, hi);
    // compute on current fragments (overlaps with outstanding loads)
    acc0 = wmma_f16(a, b0, acc0);
    acc1 = wmma_f16(a, b1, acc1);
    acc2 = wmma_f16(a, b2, acc2);
    acc3 = wmma_f16(a, b3, acc3);
    a = an; b0 = bn0; b1 = bn1; b2 = bn2; b3 = bn3;
  }
  acc0 = wmma_f16(a, b0, acc0);
  acc1 = wmma_f16(a, b1, acc1);
  acc2 = wmma_f16(a, b2, acc2);
  acc3 = wmma_f16(a, b3, acc3);

  float bv0 = 0.f, bv1 = 0.f, bv2 = 0.f, bv3 = 0.f;
  if (bias) {
    bv0 = bias[colBase + ln];
    bv1 = bias[colBase + 16 + ln];
    bv2 = bias[colBase + 32 + ln];
    bv3 = bias[colBase + 48 + ln];
  }
#pragma unroll
  for (int r = 0; r < 8; ++r) {
    int row = rowBase + r + (hi << 3);  // C/D layout: lanes>=16 hold M=r+8
    float* crow = Cout + (size_t)row * Nout + colBase;
    crow[ln]      = acc0[r] + bv0;
    crow[16 + ln] = acc1[r] + bv1;
    crow[32 + ln] = acc2[r] + bv2;
    crow[48 + ln] = acc3[r] + bv3;
  }
}

// ---------------------------------------------------------------------------
// Per-(b,h,n) layernorm + fp16 RoPE.  One wave per row, lane l owns c=l,l+32.
// Outputs: q_h (f16, BxHxNxCH), k_new/v_new (f32, BxHxNxCH)
// ---------------------------------------------------------------------------
__global__ __launch_bounds__(256) void qkv_post_kernel(
    const float* __restrict__ qkv, const float* __restrict__ qn_w,
    const float* __restrict__ qn_b, const float* __restrict__ kn_w,
    const float* __restrict__ kn_b, const int* __restrict__ xpos,
    _Float16* __restrict__ q_h, float* __restrict__ k_new,
    float* __restrict__ v_new) {
  const int gid = blockIdx.x * 8 + (threadIdx.x >> 5);  // over B*H*N = 32768
  const int lane = threadIdx.x & 31;
  const int b = gid >> 14;          // / (H*N)
  const int rem = gid & 16383;
  const int h = rem >> 10;          // / N
  const int n = rem & 1023;

  const float* rowp = qkv + (size_t)(b * N_ + n) * (3 * C_) + h * CH_;
  const int c0 = lane, c1 = lane + 32;
  float q0 = rowp[c0],          q1 = rowp[c1];
  float k0 = rowp[C_ + c0],     k1 = rowp[C_ + c1];
  float v0 = rowp[2 * C_ + c0], v1 = rowp[2 * C_ + c1];

  // layernorm over 64 elems (full-wave reduce over 32 lanes x 2 elems)
  float s = q0 + q1;
#pragma unroll
  for (int m = 1; m < 32; m <<= 1) s += __shfl_xor(s, m, 32);
  float qmu = s * (1.0f / 64.0f);
  float d0 = q0 - qmu, d1 = q1 - qmu;
  float sv = d0 * d0 + d1 * d1;
#pragma unroll
  for (int m = 1; m < 32; m <<= 1) sv += __shfl_xor(sv, m, 32);
  float qrs = rsqrtf(sv * (1.0f / 64.0f) + 1e-5f);
  q0 = d0 * qrs * qn_w[c0] + qn_b[c0];
  q1 = d1 * qrs * qn_w[c1] + qn_b[c1];

  s = k0 + k1;
#pragma unroll
  for (int m = 1; m < 32; m <<= 1) s += __shfl_xor(s, m, 32);
  float kmu = s * (1.0f / 64.0f);
  d0 = k0 - kmu; d1 = k1 - kmu;
  sv = d0 * d0 + d1 * d1;
#pragma unroll
  for (int m = 1; m < 32; m <<= 1) sv += __shfl_xor(sv, m, 32);
  float krs = rsqrtf(sv * (1.0f / 64.0f) + 1e-5f);
  k0 = d0 * krs * kn_w[c0] + kn_b[c0];
  k1 = d1 * krs * kn_w[c1] + kn_b[c1];

  // RoPE in fp16 (reference casts to fp16 before rotation)
  float pos = (float)xpos[b * N_ + n];
  // inv_freq[l] = 10000^(-l/32) = exp(-l * ln(10000)/32)
  float ang = pos * __expf(-0.28782313662425575f * (float)lane);
  _Float16 ch = (_Float16)__cosf(ang);
  _Float16 sh = (_Float16)__sinf(ang);

  _Float16 a0 = (_Float16)q0, a1 = (_Float16)q1;
  _Float16 qo0 = (_Float16)(a0 * ch - a1 * sh);
  _Float16 qo1 = (_Float16)(a1 * ch + a0 * sh);
  a0 = (_Float16)k0; a1 = (_Float16)k1;
  _Float16 ko0 = (_Float16)(a0 * ch - a1 * sh);
  _Float16 ko1 = (_Float16)(a1 * ch + a0 * sh);

  const size_t o = ((size_t)(b * H_ + h) * N_ + n) * CH_;
  q_h[o + c0] = qo0;          q_h[o + c1] = qo1;
  k_new[o + c0] = (float)ko0; k_new[o + c1] = (float)ko1;
  v_new[o + c0] = v0;         v_new[o + c1] = v1;
}

// ---------------------------------------------------------------------------
// Gather selected K/V rows. f32 rows to d_out; f16 K in (bh, key, c) layout;
// f16 V written TRANSPOSED (bh, c, key) through an LDS tile so the flash
// attention P@V B-fragments become contiguous (vectorizable to b128).
// Block = 8 waves = 8 consecutive keys of one (b,h).
// ---------------------------------------------------------------------------
__global__ __launch_bounds__(256) void select_kv_kernel(
    const float* __restrict__ past_k, const float* __restrict__ past_v,
    const float* __restrict__ k_new, const float* __restrict__ v_new,
    const int* __restrict__ idx, float* __restrict__ k_sel_f,
    float* __restrict__ v_sel_f, _Float16* __restrict__ k_sel_h,
    _Float16* __restrict__ v_selT) {
  __shared__ _Float16 vt[8][CH_ + 4];  // padded to dodge LDS bank conflicts

  const int wave = threadIdx.x >> 5;
  const int lane = threadIdx.x & 31;
  const int gid0 = blockIdx.x * 8;       // first (bh,s) of this block
  const int bh = gid0 >> 12;             // / KSEL
  const int s0 = gid0 & (KSEL_ - 1);
  const int s = s0 + wave;
  const int j = idx[s];

  const float* ksrc;
  const float* vsrc;
  if (j < SP_) {
    size_t base = ((size_t)bh * SP_ + j) * CH_;
    ksrc = past_k + base;
    vsrc = past_v + base;
  } else {
    size_t base = ((size_t)bh * N_ + (j - SP_)) * CH_;
    ksrc = k_new + base;
    vsrc = v_new + base;
  }
  const size_t o = ((size_t)bh * KSEL_ + s) * CH_;
#pragma unroll
  for (int e = 0; e < 2; ++e) {
    int c = lane + e * 32;
    float kv = ksrc[c], vv = vsrc[c];
    k_sel_f[o + c] = kv;
    v_sel_f[o + c] = vv;
    k_sel_h[o + c] = (_Float16)kv;
    vt[wave][c] = (_Float16)vv;
  }
  __syncthreads();

  // transposed write: thread t covers channel c = t>>2, key pair (t&3)*2
  const int c = threadIdx.x >> 2;
  const int g = threadIdx.x & 3;
  _Float16* dst = v_selT + ((size_t)bh * CH_ + c) * KSEL_ + s0 + 2 * g;
  dst[0] = vt[2 * g][c];
  dst[1] = vt[2 * g + 1][c];
}

// ---------------------------------------------------------------------------
// Flash attention over the 4096 selected keys. One wave = 16 query rows.
// Per 32-key block: 4 WMMAs for scores (K=64), online softmax with 16-lane
// shuffles, P re-layout through LDS, 4 WMMAs for P@V (acc 16x64).
// V is consumed from the transposed (bh, c, key) buffer -> contiguous frags.
// K fragments for the next block are prefetched before the softmax chain.
// Output f16 in (B*N, C) layout for the projection GEMM.
// ---------------------------------------------------------------------------
__global__ __launch_bounds__(256) void attn_wmma_kernel(
    const _Float16* __restrict__ q_h, const _Float16* __restrict__ k_sel_h,
    const _Float16* __restrict__ v_selT, _Float16* __restrict__ attn_h) {
  __shared__ _Float16 lds_p[8][16 * 32];  // per-wave P tile (16 rows x 32 keys)

  const int wave = threadIdx.x >> 5;
  const int lane = threadIdx.x & 31;
  const int hi = lane >> 4;
  const int ln = lane & 15;

  const int bh = blockIdx.x >> 3;      // B*H = 32
  const int nchunk = blockIdx.x & 7;   // N/128 = 8
  const int n0 = nchunk * 128 + wave * 16;
  const int b = bh >> 4, h = bh & 15;

  const _Float16* qbase = q_h + ((size_t)bh * N_ + n0) * CH_;
  const _Float16* kbase = k_sel_h + (size_t)bh * KSEL_ * CH_;
  const _Float16* vtbase = v_selT + (size_t)bh * CH_ * KSEL_;

  // Q tile (16x64) as two A-fragments, resident for the whole loop
  const v16h qa0 = load_a_frag(qbase, CH_, ln, 0, hi);
  const v16h qa1 = load_a_frag(qbase, CH_, ln, 32, hi);

  v8f acc[4] = {};
  float mrow[8], lrow[8];
#pragma unroll
  for (int r = 0; r < 8; ++r) { mrow[r] = -__builtin_inff(); lrow[r] = 0.f; }

  const float scale = 0.125f;  // CH^-0.5

  // preload first K block fragments
  v16h kb00 = load_b_frag(kbase, CH_, ln, 0, hi);
  v16h kb01 = load_b_frag(kbase, CH_, ln, 32, hi);
  v16h kb10 = load_b_frag(kbase, CH_, 16 + ln, 0, hi);
  v16h kb11 = load_b_frag(kbase, CH_, 16 + ln, 32, hi);

  for (int ks = 0; ks < KSEL_; ks += 32) {
    // scores S(16q x 32keys) = Q @ Ksel^T  (contraction over c=64)
    v8f s0 = {}, s1 = {};
    s0 = wmma_f16(qa0, kb00, s0);
    s0 = wmma_f16(qa1, kb01, s0);
    s1 = wmma_f16(qa0, kb10, s1);
    s1 = wmma_f16(qa1, kb11, s1);

    // V fragments for this block (transposed layout -> contiguous pairs)
    v16h vb0 = load_b_frag(vtbase, KSEL_, ln, ks, hi);
    v16h vb1 = load_b_frag(vtbase, KSEL_, 16 + ln, ks, hi);
    v16h vb2 = load_b_frag(vtbase, KSEL_, 32 + ln, ks, hi);
    v16h vb3 = load_b_frag(vtbase, KSEL_, 48 + ln, ks, hi);

    // prefetch next K block while the softmax VALU chain runs
    if (ks + 32 < KSEL_) {
      kb00 = load_b_frag(kbase, CH_, ks + 32 + ln, 0, hi);
      kb01 = load_b_frag(kbase, CH_, ks + 32 + ln, 32, hi);
      kb10 = load_b_frag(kbase, CH_, ks + 48 + ln, 0, hi);
      kb11 = load_b_frag(kbase, CH_, ks + 48 + ln, 32, hi);
    }

    // online softmax; row M lives across one 16-lane group
#pragma unroll
    for (int r = 0; r < 8; ++r) {
      float a0 = s0[r] * scale;
      float a1 = s1[r] * scale;
      float mx = fmaxf(a0, a1);
#pragma unroll
      for (int m = 1; m < 16; m <<= 1) mx = fmaxf(mx, __shfl_xor(mx, m, 32));
      float mnew = fmaxf(mrow[r], mx);
      float p0 = __expf(a0 - mnew);
      float p1 = __expf(a1 - mnew);
      float rs = p0 + p1;
#pragma unroll
      for (int m = 1; m < 16; m <<= 1) rs += __shfl_xor(rs, m, 32);
      float corr = __expf(mrow[r] - mnew);
      lrow[r] = lrow[r] * corr + rs;
      mrow[r] = mnew;
#pragma unroll
      for (int t = 0; t < 4; ++t) acc[t][r] *= corr;
      const int row = r + (hi << 3);
      lds_p[wave][row * 32 + ln]      = (_Float16)p0;
      lds_p[wave][row * 32 + 16 + ln] = (_Float16)p1;
    }
    // wave-local LDS exchange: wait for our own DS stores before re-reading
    asm volatile("s_wait_dscnt 0" ::: "memory");

    // P as A-fragment (16x32, contraction over keys)
    v16h pa;
#pragma unroll
    for (int j = 0; j < 8; ++j) {
      int k = kmapA(j, hi);
      pa[2 * j]     = lds_p[wave][ln * 32 + k];
      pa[2 * j + 1] = lds_p[wave][ln * 32 + k + 1];
    }

    // acc(16x64) += P @ V(32keys x 64c)
    acc[0] = wmma_f16(pa, vb0, acc[0]);
    acc[1] = wmma_f16(pa, vb1, acc[1]);
    acc[2] = wmma_f16(pa, vb2, acc[2]);
    acc[3] = wmma_f16(pa, vb3, acc[3]);
  }

  // epilogue: normalize and store f16 in (B*N, C) layout
#pragma unroll
  for (int r = 0; r < 8; ++r) {
    float inv = 1.0f / lrow[r];
    int row = n0 + r + (hi << 3);
    _Float16* orow = attn_h + ((size_t)(b * N_ + row)) * C_ + h * CH_;
#pragma unroll
    for (int t = 0; t < 4; ++t)
      orow[t * 16 + ln] = (_Float16)(acc[t][r] * inv);
  }
}

// ---------------------------------------------------------------------------
// Host launcher
// ---------------------------------------------------------------------------
extern "C" void kernel_launch(void* const* d_in, const int* in_sizes, int n_in,
                              void* d_out, int out_size, void* d_ws,
                              size_t ws_size, hipStream_t stream) {
  (void)in_sizes; (void)n_in; (void)out_size; (void)ws_size;

  const float* x      = (const float*)d_in[0];
  const float* Wqkv   = (const float*)d_in[1];
  const float* Wproj  = (const float*)d_in[2];
  const float* bproj  = (const float*)d_in[3];
  const float* qn_w   = (const float*)d_in[4];
  const float* qn_b   = (const float*)d_in[5];
  const float* kn_w   = (const float*)d_in[6];
  const float* kn_b   = (const float*)d_in[7];
  const float* past_k = (const float*)d_in[8];
  const float* past_v = (const float*)d_in[9];
  const int*   xpos   = (const int*)d_in[10];
  const int*   idx    = (const int*)d_in[11];

  // d_out = [out (B*N*C f32) | k_sel (B*H*KSEL*CH f32) | v_sel (same)]
  float* out_f   = (float*)d_out;
  float* k_sel_f = out_f + (size_t)B_ * N_ * C_;
  float* v_sel_f = k_sel_f + (size_t)B_ * H_ * KSEL_ * CH_;

  // workspace carve-up
  char* ws = (char*)d_ws;
  size_t off = 0;
  auto carve = [&](size_t bytes) -> void* {
    void* p = ws + off;
    off = (off + bytes + 255) & ~(size_t)255;
    return p;
  };
  const size_t nx    = (size_t)B_ * N_ * C_;           // 2M
  const size_t nwqkv = (size_t)3 * C_ * C_;            // 3M
  const size_t nwprj = (size_t)C_ * C_;                // 1M
  const size_t nbhnc = (size_t)B_ * H_ * N_ * CH_;     // 2M
  const size_t nsel  = (size_t)B_ * H_ * KSEL_ * CH_;  // 8M

  _Float16* x_h     = (_Float16*)carve(nx * 2);
  _Float16* wqkv_h  = (_Float16*)carve(nwqkv * 2);
  _Float16* wproj_h = (_Float16*)carve(nwprj * 2);
  float*    qkv_f   = (float*)carve((size_t)(B_ * N_) * 3 * C_ * 4);
  _Float16* q_h     = (_Float16*)carve(nbhnc * 2);
  float*    k_new   = (float*)carve(nbhnc * 4);
  float*    v_new   = (float*)carve(nbhnc * 4);
  _Float16* k_sel_h = (_Float16*)carve(nsel * 2);
  _Float16* v_selT  = (_Float16*)carve(nsel * 2);  // (bh, c, key) transposed
  _Float16* attn_h  = (_Float16*)carve(nx * 2);

  // 1) f32 -> f16 converts
  f32_to_f16_kernel<<<1024, 256, 0, stream>>>(x, x_h, (int)nx);
  f32_to_f16_kernel<<<1024, 256, 0, stream>>>(Wqkv, wqkv_h, (int)nwqkv);
  f32_to_f16_kernel<<<512, 256, 0, stream>>>(Wproj, wproj_h, (int)nwprj);

  // 2) QKV GEMM: (2048x1024) @ (3072x1024)^T -> qkv_f
  {
    int M = B_ * N_, Nout = 3 * C_, K = C_;
    int waves = (M / 16) * (Nout / 64);  // 6144
    gemm_wmma_kernel<<<waves / 8, 256, 0, stream>>>(x_h, wqkv_h, nullptr,
                                                    qkv_f, M, Nout, K);
  }

  // 3) layernorm + RoPE -> q_h (f16), k_new/v_new (f32)
  qkv_post_kernel<<<(B_ * H_ * N_) / 8, 256, 0, stream>>>(
      qkv_f, qn_w, qn_b, kn_w, kn_b, xpos, q_h, k_new, v_new);

  // 4) gather selected K/V (f32 outputs + f16 working copies, V transposed)
  select_kv_kernel<<<(B_ * H_ * KSEL_) / 8, 256, 0, stream>>>(
      past_k, past_v, k_new, v_new, idx, k_sel_f, v_sel_f, k_sel_h, v_selT);

  // 5) flash attention over 4096 selected keys -> attn_h (f16, (B*N, C))
  attn_wmma_kernel<<<(B_ * H_) * (N_ / 128), 256, 0, stream>>>(
      q_h, k_sel_h, v_selT, attn_h);

  // 6) output projection: (2048x1024) @ (1024x1024)^T + bproj -> d_out
  {
    int M = B_ * N_, Nout = C_, K = C_;
    int waves = (M / 16) * (Nout / 64);  // 2048
    gemm_wmma_kernel<<<waves / 8, 256, 0, stream>>>(attn_h, wproj_h, bproj,
                                                    out_f, M, Nout, K);
  }
}
